// LSTM_43980465111351
// MI455X (gfx1250) — matrix operand order repaired and checked by reading, not compiled, not run on hardware
//
#include <hip/hip_runtime.h>
#include <stdint.h>

// ---------------------------------------------------------------------------
// Types for CDNA5 WMMA (wave32): bf16 A/B fragments (8 VGPRs), f32 C/D (8 VGPRs)
// ---------------------------------------------------------------------------
typedef __attribute__((ext_vector_type(16))) __bf16 v16bf;
typedef __attribute__((ext_vector_type(8)))  float  v8f;

union FragU {
  uint4 q[2];
  v16bf v;
};

__device__ __forceinline__ uint16_t f2bfu(float f) {
  union { __bf16 h; uint16_t u; } r;
  r.h = (__bf16)f;
  return r.u;
}
__device__ __forceinline__ uint32_t pk2(float a, float b) {
  union { __bf16 h[2]; uint32_t u; } r;
  r.h[0] = (__bf16)a;
  r.h[1] = (__bf16)b;
  return r.u;
}
__device__ __forceinline__ float sigmoidf_(float x) {
  return 1.0f / (1.0f + __expf(-x));
}

// A fragment from a bf16 row (row base already includes the per-lane ksel).
// 16-bit A layout (16x32): lanes 0-15 hold M=lane, K=k0+0..7 / k0+16..23;
// lanes 16-31 hold M=lane-16, K=k0+8..15 / k0+24..31.
__device__ __forceinline__ v16bf loadAfrag(const uint16_t* row, int koff) {
  const uint16_t* p = row + koff;
  FragU f;
  f.q[0] = *(const uint4*)p;
  f.q[1] = *(const uint4*)(p + 16);
  return f.v;
}

// 4 gate B fragments from one uniform base (gate-interleaved prepack layout):
// gate g at +g*512 halves; 8 b128 loads, single saddr, immediate offsets.
__device__ __forceinline__ void loadB4(const uint16_t* uni, int laneHalfOff,
                                       v16bf& B0, v16bf& B1, v16bf& B2, v16bf& B3) {
  const uint16_t* p = uni + laneHalfOff;
  FragU f0, f1, f2, f3;
  f0.q[0] = *(const uint4*)(p);        f0.q[1] = *(const uint4*)(p + 8);
  f1.q[0] = *(const uint4*)(p + 512);  f1.q[1] = *(const uint4*)(p + 520);
  f2.q[0] = *(const uint4*)(p + 1024); f2.q[1] = *(const uint4*)(p + 1032);
  f3.q[0] = *(const uint4*)(p + 1536); f3.q[1] = *(const uint4*)(p + 1544);
  B0 = f0.v; B1 = f1.v; B2 = f2.v; B3 = f3.v;
}

__device__ __forceinline__ void mma_set(v8f& a0, v8f& a1, v8f& a2, v8f& a3,
                                        v16bf A, v16bf B0, v16bf B1, v16bf B2,
                                        v16bf B3) {
  a0 = __builtin_amdgcn_wmma_f32_16x16x32_bf16(false, A, false, B0, (short)0, a0, false, false);
  a1 = __builtin_amdgcn_wmma_f32_16x16x32_bf16(false, A, false, B1, (short)0, a1, false, false);
  a2 = __builtin_amdgcn_wmma_f32_16x16x32_bf16(false, A, false, B2, (short)0, a2, false, false);
  a3 = __builtin_amdgcn_wmma_f32_16x16x32_bf16(false, A, false, B3, (short)0, a3, false, false);
}

// ---------------------------------------------------------------------------
// One LSTM cell step for one wave: 16 hidden cols (jt), 16 batch rows (mt),
// all 4 gates. K = 2048 = [low 1024 | high 1024], both bf16 sources.
// Branch-free ping-pong pipeline, carried across the low/high boundary.
// ---------------------------------------------------------------------------
__device__ __forceinline__ void lstm_cell(
    int lane, int jt, int mt,
    const uint16_t* aLow,   // this wave's row base (+ksel) for K 0..1023
    const uint16_t* aHigh,  // this wave's row base (+ksel) for K 1024..2047
    const uint16_t* wUni,   // uniform weight base for this jt
    float bf_, float bi_, float bg_, float bo_,
    float* cst,             // [8] cell state in registers
    uint16_t* curH,         // this layer's current h out
    uint16_t* h1all)        // layer1: history slab; layer0: nullptr
{
  const int lo16 = lane << 4;
  v8f a0 = (v8f){0, 0, 0, 0, 0, 0, 0, 0};
  v8f a1 = a0, a2 = a0, a3 = a0;

  v16bf A0, B00, B10, B20, B30;  // set 0
  v16bf A1, B01, B11, B21, B31;  // set 1

  A0 = loadAfrag(aLow, 0);
  loadB4(wUni, lo16, B00, B10, B20, B30);
#pragma unroll 1
  for (int kt = 0; kt < 30; kt += 2) {
    A1 = loadAfrag(aLow, (kt + 1) * 32);
    loadB4(wUni + (kt + 1) * 2048, lo16, B01, B11, B21, B31);
    mma_set(a0, a1, a2, a3, A0, B00, B10, B20, B30);       // kt
    A0 = loadAfrag(aLow, (kt + 2) * 32);
    loadB4(wUni + (kt + 2) * 2048, lo16, B00, B10, B20, B30);
    mma_set(a0, a1, a2, a3, A1, B01, B11, B21, B31);       // kt+1
  }
  A1 = loadAfrag(aLow, 31 * 32);
  loadB4(wUni + 31 * 2048, lo16, B01, B11, B21, B31);
  mma_set(a0, a1, a2, a3, A0, B00, B10, B20, B30);         // kt = 30
  A0 = loadAfrag(aHigh, 0);
  loadB4(wUni + 32 * 2048, lo16, B00, B10, B20, B30);
  mma_set(a0, a1, a2, a3, A1, B01, B11, B21, B31);         // kt = 31
#pragma unroll 1
  for (int kt = 32; kt < 62; kt += 2) {
    A1 = loadAfrag(aHigh, (kt - 31) * 32);
    loadB4(wUni + (kt + 1) * 2048, lo16, B01, B11, B21, B31);
    mma_set(a0, a1, a2, a3, A0, B00, B10, B20, B30);       // kt
    A0 = loadAfrag(aHigh, (kt - 30) * 32);
    loadB4(wUni + (kt + 2) * 2048, lo16, B00, B10, B20, B30);
    mma_set(a0, a1, a2, a3, A1, B01, B11, B21, B31);       // kt+1
  }
  A1 = loadAfrag(aHigh, 31 * 32);
  loadB4(wUni + 63 * 2048, lo16, B01, B11, B21, B31);
  mma_set(a0, a1, a2, a3, A0, B00, B10, B20, B30);         // kt = 62
  mma_set(a0, a1, a2, a3, A1, B01, B11, B21, B31);         // kt = 63

  // Elementwise LSTM cell. C/D layout: VGPR r -> M = r (+8 for lanes 16-31),
  // N = jt*16 + (lane&15).
  const int lhalf = lane & 15;
  const int n = jt * 16 + lhalf;
#pragma unroll
  for (int r = 0; r < 8; ++r) {
    const int m = mt * 16 + r + ((lane & 16) ? 8 : 0);
    float f = a0[r] + bf_;
    float ii = a1[r] + bi_;
    float g = a2[r] + bg_;
    float o = a3[r] + bo_;
    float cp = cst[r];
    float cn = sigmoidf_(f) * cp + sigmoidf_(ii) * tanhf(g);
    cst[r] = cn;
    float h = sigmoidf_(o) * tanhf(cn);
    uint16_t hb = f2bfu(h);
    curH[m * 1024 + n] = hb;
    if (h1all) h1all[m * 1024 + n] = hb;
  }
}

// Device-wide barrier: release + arrive, spin with s_sleep, acquire.
__device__ __forceinline__ void grid_barrier(unsigned* ctr, unsigned nblk) {
  __threadfence();
  __syncthreads();
  if (threadIdx.x == 0) {
    __hip_atomic_fetch_add(ctr, 1u, __ATOMIC_RELEASE, __HIP_MEMORY_SCOPE_AGENT);
    while (__hip_atomic_load(ctr, __ATOMIC_RELAXED, __HIP_MEMORY_SCOPE_AGENT) < nblk)
      __builtin_amdgcn_s_sleep(1);
  }
  __syncthreads();
  __threadfence();
}

// ---------------------------------------------------------------------------
// Persistent recurrent kernel: 32 blocks x 128 threads = 128 waves
// (1 wave/SIMD -> VGPR allocation always fits; pipeline hides latency).
// Wave gw -> (jt = gw>>1, mt = gw&1). Cell states stay in registers.
// Fused phases: L0(0) | {L1(t-1), L0(t)} x 511 | L1(511); one barrier/phase.
// ---------------------------------------------------------------------------
__global__ void __launch_bounds__(128, 1) lstm_persistent(
    const uint16_t* __restrict__ Xbf, const float* __restrict__ b0,
    const float* __restrict__ b1, const uint16_t* __restrict__ W0p,
    const uint16_t* __restrict__ W1p, uint16_t* __restrict__ H0,
    uint16_t* __restrict__ H1, uint16_t* __restrict__ H1all,
    unsigned* __restrict__ ctrs) {
  const int tid = threadIdx.x;
  const int lane = tid & 31;
  const int gw = (int)((blockIdx.x * blockDim.x + tid) >> 5);        // 0..127
  const int jt = __builtin_amdgcn_readfirstlane(gw >> 1);            // 0..63
  const int mt = __builtin_amdgcn_readfirstlane(gw & 1);             // 0..1
  const int lhalf = lane & 15;
  const int ksel = (lane & 16) ? 8 : 0;
  const int mrow = mt * 16 + lhalf;
  const int n = jt * 16 + lhalf;

  const float b0f = b0[n], b0i = b0[n + 1024], b0g = b0[n + 2048], b0o = b0[n + 3072];
  const float b1f = b1[n], b1i = b1[n + 1024], b1g = b1[n + 2048], b1o = b1[n + 3072];

  const uint16_t* wU0 = W0p + ((size_t)jt << 17);  // 256 tiles * 512 halves per jt
  const uint16_t* wU1 = W1p + ((size_t)jt << 17);
  const uint16_t* aX = Xbf + (size_t)mrow * 512 * 1024 + ksel;   // + t*1024 per step
  const int rofs = mrow * 1024 + ksel;

  float c0[8], c1[8];
#pragma unroll
  for (int i = 0; i < 8; ++i) { c0[i] = 0.0f; c1[i] = 0.0f; }

  // Phase 0: L0(t=0). Reads x(0), h0_prev = H0[1] (zeroed); writes H0[0].
  lstm_cell(lane, jt, mt, aX, H0 + 32768 + rofs, wU0,
            b0f, b0i, b0g, b0o, c0, H0, nullptr);
  grid_barrier(&ctrs[0], 32);

#pragma unroll 1
  for (int t = 1; t < 512; ++t) {
    const int cur = (t & 1) * 32768;   // parity of t
    const int prv = 32768 - cur;       // parity of t-1
    // L1(t-1): combined = [h0(t-1) | h1(t-2)]; h0(t-1) in H0[prv], h1(t-2) in H1[cur].
    lstm_cell(lane, jt, mt, H0 + prv + rofs, H1 + cur + rofs, wU1,
              b1f, b1i, b1g, b1o, c1, H1 + prv,
              H1all + (size_t)(t - 1) * 32768);
    // L0(t): combined = [x(t) | h0(t-1)]; writes H0[cur].
    lstm_cell(lane, jt, mt, aX + (size_t)t * 1024, H0 + prv + rofs, wU0,
              b0f, b0i, b0g, b0o, c0, H0 + cur, nullptr);
    grid_barrier(&ctrs[t], 32);
  }
  // Final: L1(t=511). h0(511) in H0[1], h1(510) in H1[0]; writes H1[1].
  lstm_cell(lane, jt, mt, H0 + 32768 + rofs, H1 + rofs, wU1,
            b1f, b1i, b1g, b1o, c1, H1 + 32768,
            H1all + (size_t)511 * 32768);
}

// ---------------------------------------------------------------------------
// Weight prepack: fp32 [N][K] row-major -> bf16 WMMA-B tiles, gate-interleaved.
// tile = ((ngi*KT + kt)*4 + q); covered N-tile = q*qStride + ngi*gStride.
// Tile = 1KB: lane L stores W[nt*16 + L%16][kt*32 + (L<16?0:8) + {0..7, 16..23}].
// ---------------------------------------------------------------------------
__global__ void prepack_w(const float* __restrict__ W, uint16_t* __restrict__ out,
                          int NG, int KT, int K, int qStride, int gStride) {
  const int id = (int)(blockIdx.x * blockDim.x + threadIdx.x);
  if (id >= NG * KT * 4 * 32) return;
  const int lane = id & 31;
  const int tile = id >> 5;
  const int q = tile & 3;
  const int rem = tile >> 2;
  const int kt = rem % KT;
  const int ngi = rem / KT;
  const int ntg = q * qStride + ngi * gStride;
  const int nrow = ntg * 16 + (lane & 15);
  const int kstart = kt * 32 + ((lane & 16) ? 8 : 0);
  const float* src = W + (size_t)nrow * K + kstart;
  uint32_t w[8];
#pragma unroll
  for (int j = 0; j < 8; ++j) w[j] = pk2(src[2 * j + (j >= 4 ? 8 : 0)], src[2 * j + 1 + (j >= 4 ? 8 : 0)]);
  uint4* dst = (uint4*)(out + ((size_t)tile << 9) + (lane << 4));
  dst[0] = make_uint4(w[0], w[1], w[2], w[3]);
  dst[1] = make_uint4(w[4], w[5], w[6], w[7]);
}

// Convert x (fp32, [32][512][1024]) to bf16, same layout.
__global__ void cvt_x(const float* __restrict__ x, uint16_t* __restrict__ xb) {
  const int id = (int)(blockIdx.x * blockDim.x + threadIdx.x);  // 4 elems each
  const float4 v = ((const float4*)x)[id];
  ((uint2*)xb)[id] = make_uint2(pk2(v.x, v.y), pk2(v.z, v.w));
}

// Zero h double-buffers and barrier counters (re-run every graph replay).
__global__ void init_state(uint16_t* H0, uint16_t* H1, unsigned* ctrs) {
  const int id = (int)(blockIdx.x * blockDim.x + threadIdx.x);
  if (id < 65536) { H0[id] = 0; H1[id] = 0; }
  if (id < 1024) ctrs[id] = 0;
}

// ---------------------------------------------------------------------------
// Projection: out[b][t][o] = h1[t*32+b][:] . Wd[o][:] + bd[o]
// Wave task: 16 rows (mtile) x 64 cols (ng: 4 interleaved tiles), K = 1024.
// ---------------------------------------------------------------------------
__global__ void __launch_bounds__(256) proj_kernel(
    const uint16_t* __restrict__ H1all, const uint16_t* __restrict__ Wdp,
    const float* __restrict__ bd, float* __restrict__ out) {
  const int lane = threadIdx.x & 31;
  const int gw = (int)((blockIdx.x * blockDim.x + threadIdx.x) >> 5);  // 0..16383
  const int mtile = __builtin_amdgcn_readfirstlane(gw >> 4);  // 0..1023
  const int ng = __builtin_amdgcn_readfirstlane(gw & 15);     // 0..15
  const int lhalf = lane & 15;
  const int ksel = (lane & 16) ? 8 : 0;
  const int lo16 = lane << 4;
  const uint16_t* arow = H1all + (size_t)(mtile * 16 + lhalf) * 1024 + ksel;
  const uint16_t* wU = Wdp + ((size_t)ng << 16);  // 128 tiles * 512 halves per ng

  v8f a0 = (v8f){0, 0, 0, 0, 0, 0, 0, 0};
  v8f a1 = a0, a2 = a0, a3 = a0;
  float bias[4];
#pragma unroll
  for (int g = 0; g < 4; ++g) bias[g] = bd[(ng * 4 + g) * 16 + lhalf];

#pragma unroll 1
  for (int kt = 0; kt < 32; ++kt) {
    v16bf A = loadAfrag(arow, kt * 32);
    v16bf B0, B1, B2, B3;
    loadB4(wU + kt * 2048, lo16, B0, B1, B2, B3);
    mma_set(a0, a1, a2, a3, A, B0, B1, B2, B3);
  }
  const v8f accs[4] = {a0, a1, a2, a3};
#pragma unroll
  for (int g = 0; g < 4; ++g) {
    const int nO = (ng * 4 + g) * 16 + lhalf;
#pragma unroll
    for (int r = 0; r < 8; ++r) {
      const int R = mtile * 16 + r + ((lane & 16) ? 8 : 0);  // flat row = t*32 + b
      const int tt = R >> 5;
      const int m = R & 31;
      out[((size_t)m * 512 + tt) * 1024 + nO] = accs[g][r] + bias[g];
    }
  }
}

// ---------------------------------------------------------------------------
// Launch
// ---------------------------------------------------------------------------
extern "C" void kernel_launch(void* const* d_in, const int* in_sizes, int n_in,
                              void* d_out, int out_size, void* d_ws, size_t ws_size,
                              hipStream_t stream) {
  const float* x  = (const float*)d_in[0];
  const float* W0 = (const float*)d_in[1];
  const float* b0 = (const float*)d_in[2];
  const float* W1 = (const float*)d_in[3];
  const float* b1 = (const float*)d_in[4];
  const float* Wd = (const float*)d_in[5];
  const float* bd = (const float*)d_in[6];
  float* out = (float*)d_out;

  char* ws = (char*)d_ws;
  uint16_t* W0p   = (uint16_t*)(ws);                                   // 16 MB
  uint16_t* W1p   = (uint16_t*)(ws + (size_t)(16u << 20));             // 16 MB
  uint16_t* Wdp   = (uint16_t*)(ws + (size_t)(32u << 20));             //  2 MB
  uint16_t* H1all = (uint16_t*)(ws + (size_t)(34u << 20));             // 32 MB
  uint16_t* Xbf   = (uint16_t*)(ws + (size_t)(66u << 20));             // 32 MB
  uint16_t* H0    = (uint16_t*)(ws + (size_t)(98u << 20));             // 128 KB
  uint16_t* H1    = (uint16_t*)(ws + (size_t)(98u << 20) + (128u << 10));
  unsigned* ctrs  = (unsigned*)(ws + (size_t)(98u << 20) + (256u << 10));

  init_state<<<256, 256, 0, stream>>>(H0, H1, ctrs);
  cvt_x<<<16384, 256, 0, stream>>>(x, Xbf);
  prepack_w<<<2048, 256, 0, stream>>>(W0, W0p, 64, 64, 2048, 64, 1);
  prepack_w<<<2048, 256, 0, stream>>>(W1, W1p, 64, 64, 2048, 64, 1);
  prepack_w<<<256, 256, 0, stream>>>(Wd, Wdp, 16, 32, 1024, 1, 4);
  lstm_persistent<<<32, 128, 0, stream>>>(Xbf, b0, b1, W0p, W1p, H0, H1, H1all, ctrs);
  proj_kernel<<<2048, 256, 0, stream>>>(H1all, Wdp, bd, out);
}
